// MultiHeadedAttentionWithRelativePosition_39367670235445
// MI455X (gfx1250) — compile-verified, matrix operand order checked by reading, not run on hardware
//
#include <hip/hip_runtime.h>
#include <hip/hip_bf16.h>
#include <math.h>

// ---------------------------------------------------------------------------
// MHA w/ relative position bias, MI455X (gfx1250, wave32, WMMA bf16).
// B=2 S=2048 D=1024 H=16 DK=64 WIDTH=64, left-to-right band, full softmax.
// Pipeline:
//   wprep (fp32 W[K,N] -> bf16 Wt[N,K], LDS-tiled transpose)
//   3x proj GEMM (128x128 tiles, bf16 WMMA; V written dk-major [b,h,dk,s])
//   flash attention (64-query x 64-key tiles, online softmax)
//   output GEMM (bf16 -> fp32)
// Tile staging uses GLOBAL_LOAD_ASYNC_TO_LDS_B128 when available.
// ---------------------------------------------------------------------------

constexpr int Bc  = 2;
constexpr int Sc  = 2048;
constexpr int Dc  = 1024;
constexpr int Hc  = 16;
constexpr int DKc = 64;
constexpr int WIDTHc = 64;
#define SCALE_F 0.125f   /* 1/sqrt(64) */

typedef __attribute__((ext_vector_type(16))) __bf16 v16bf;
typedef __attribute__((ext_vector_type(8)))  __bf16 v8bf;
typedef __attribute__((ext_vector_type(4)))  __bf16 v4bf;
typedef __attribute__((ext_vector_type(8)))  float  v8f;
typedef __attribute__((ext_vector_type(4)))  int    v4i;

#if defined(__has_builtin)
#if __has_builtin(__builtin_amdgcn_global_load_async_to_lds_b128)
#define ASYNC_COPY_OK 1
#endif
#endif

// 16-byte global -> LDS copy; async (ASYNCcnt) path on CDNA5 when declared.
// Builtin signature (from clang diagnostic): param0 = v4i addrspace(1)*.
__device__ __forceinline__ void copy16_to_lds(const __bf16* g, __bf16* l) {
#ifdef ASYNC_COPY_OK
  __builtin_amdgcn_global_load_async_to_lds_b128(
      (__attribute__((address_space(1))) v4i*)(g),
      (__attribute__((address_space(3))) v4i*)(l), 0, 0);
#else
  *(uint4*)l = *(const uint4*)g;
#endif
}

__device__ __forceinline__ void async_copy_wait() {
#ifdef ASYNC_COPY_OK
  asm volatile("s_wait_asynccnt 0" ::: "memory");
#endif
}

__device__ __forceinline__ v16bf ld_frag(const __bf16* p0, const __bf16* p1) {
  v8bf lo = *(const v8bf*)p0;
  v8bf hi = *(const v8bf*)p1;
  v16bf r;
#pragma unroll
  for (int i = 0; i < 8; ++i) { r[i] = lo[i]; r[i + 8] = hi[i]; }
  return r;
}

__device__ __forceinline__ v8f v8f_zero() {
  v8f z = {0.f, 0.f, 0.f, 0.f, 0.f, 0.f, 0.f, 0.f};
  return z;
}

// ---------------------------------------------------------------------------
// Weight prep: W fp32 [K=1024][N=1024] -> Wt bf16 [N][K] (LDS-tiled transpose)
// ---------------------------------------------------------------------------
__global__ __launch_bounds__(256) void wprep_kernel(const float* __restrict__ W,
                                                    __bf16* __restrict__ Wt) {
  __shared__ float tile[32][33];
  const int tid = threadIdx.x;
  const int k0 = blockIdx.x * 32;
  const int n0 = blockIdx.y * 32;
#pragma unroll
  for (int i = 0; i < 4; ++i) {
    const int e = tid + i * 256;
    const int r = e >> 5, c = e & 31;            // r = k, c = n (coalesced read)
    tile[r][c] = W[(size_t)(k0 + r) * Dc + n0 + c];
  }
  __syncthreads();
#pragma unroll
  for (int i = 0; i < 4; ++i) {
    const int e = tid + i * 256;
    const int r = e >> 5, c = e & 31;            // r = n, c = k (coalesced write)
    Wt[(size_t)(n0 + r) * Dc + k0 + c] = (__bf16)tile[c][r];
  }
}

// ---------------------------------------------------------------------------
// Tiled GEMM: Y = X[M,K] @ W[K,N] + bias, with W given as bf16 Wt[N][K].
// M=4096, K=N=1024.  Block = 256 threads (8 waves), tile 128x128, BK=32.
// Waves in 2x4 grid; each wave: 64 rows x 32 cols = 4 A-frags x 2 B-frags
// -> 8 WMMAs per K-step from 12 ds_load_b128.
// IN_BF16:  X is bf16 (workspace) else fp32 (harness input).
// OUT_MODE: 0 = bf16 [b,h,s,dk], 1 = bf16 [b,h,dk,s], 2 = fp32 flat [m,n].
// ---------------------------------------------------------------------------
template <bool IN_BF16, int OUT_MODE>
__global__ __launch_bounds__(256) void gemm_kernel(const void* __restrict__ Xv,
                                                   const __bf16* __restrict__ Wt,
                                                   const float* __restrict__ bias,
                                                   void* __restrict__ Yv) {
  constexpr int K = Dc;
  constexpr int N = Dc;

  __shared__ __align__(16) __bf16 lA[128 * 32];   // [row][k]
  __shared__ __align__(16) __bf16 lB[128 * 32];   // [n][k]

  const int tid   = threadIdx.x;
  const int lane  = tid & 31;
  const int wv    = tid >> 5;              // 0..7
  const int lh    = lane & 15;
  const int khalf = lane >> 4;
  const int nBase = blockIdx.x * 128;
  const int mBase = blockIdx.y * 128;
  const int rbase = (wv >> 2) * 64;        // wave row offset within tile
  const int cbase = (wv & 3) * 32;         // wave col offset within tile

  const float*  Xf = (const float*)Xv;
  const __bf16* Xb = (const __bf16*)Xv;

  v8f acc[4][2];
#pragma unroll
  for (int a = 0; a < 4; ++a)
#pragma unroll
    for (int bi = 0; bi < 2; ++bi) acc[a][bi] = v8f_zero();

  for (int k0 = 0; k0 < K; k0 += 32) {
    // ---- stage A tile 128x32 ----
    if (IN_BF16) {
#pragma unroll
      for (int i = 0; i < 2; ++i) {
        const int chunk = tid + i * 256;             // 512 chunks of 8 bf16
        const int r = chunk >> 2, c8 = (chunk & 3) * 8;
        copy16_to_lds(&Xb[(size_t)(mBase + r) * K + k0 + c8], &lA[r * 32 + c8]);
      }
    } else {
#pragma unroll
      for (int i = 0; i < 4; ++i) {
        const int chunk = tid + i * 256;             // 1024 chunks of 4 fp32
        const int r = chunk >> 3, c4 = (chunk & 7) * 4;
        const float4 xv = *(const float4*)&Xf[(size_t)(mBase + r) * K + k0 + c4];
        v4bf pk;
        pk[0] = (__bf16)xv.x; pk[1] = (__bf16)xv.y;
        pk[2] = (__bf16)xv.z; pk[3] = (__bf16)xv.w;
        *(v4bf*)&lA[r * 32 + c4] = pk;
      }
    }
    // ---- stage B tile 128x32 from Wt[n][k] ----
#pragma unroll
    for (int i = 0; i < 2; ++i) {
      const int chunk = tid + i * 256;               // 512 chunks of 8 bf16
      const int n = chunk >> 2, c8 = (chunk & 3) * 8;
      copy16_to_lds(&Wt[(size_t)(nBase + n) * K + k0 + c8], &lB[n * 32 + c8]);
    }
    async_copy_wait();
    __syncthreads();

    v16bf af[4], bf[2];
#pragma unroll
    for (int a = 0; a < 4; ++a) {
      const int arow = rbase + a * 16 + lh;
      af[a] = ld_frag(&lA[arow * 32 + khalf * 8],
                      &lA[arow * 32 + 16 + khalf * 8]);
    }
#pragma unroll
    for (int bi = 0; bi < 2; ++bi) {
      const int bcol = cbase + bi * 16 + lh;
      bf[bi] = ld_frag(&lB[bcol * 32 + khalf * 8],
                       &lB[bcol * 32 + 16 + khalf * 8]);
    }
#pragma unroll
    for (int a = 0; a < 4; ++a)
#pragma unroll
      for (int bi = 0; bi < 2; ++bi)
        acc[a][bi] = __builtin_amdgcn_wmma_f32_16x16x32_bf16(
            false, af[a], false, bf[bi], (short)0, acc[a][bi], false, false);
    __syncthreads();
  }

  // ---- epilogue: bias + store ----
#pragma unroll
  for (int a = 0; a < 4; ++a) {
#pragma unroll
    for (int bi = 0; bi < 2; ++bi) {
      const int ncol = nBase + cbase + bi * 16 + lh;
      const float bv = bias[ncol];
#pragma unroll
      for (int r = 0; r < 8; ++r) {
        const int mrow = mBase + rbase + a * 16 + r + khalf * 8;
        const float val = acc[a][bi][r] + bv;
        const int h = ncol >> 6, d = ncol & 63;
        const int bb = mrow >> 11, s = mrow & (Sc - 1);
        if (OUT_MODE == 0) {
          ((__bf16*)Yv)[(((size_t)(bb * Hc + h) * Sc + s) * DKc) + d] = (__bf16)val;
        } else if (OUT_MODE == 1) {
          ((__bf16*)Yv)[(((size_t)(bb * Hc + h) * DKc + d) * Sc) + s] = (__bf16)val;
        } else {
          ((float*)Yv)[(size_t)mrow * N + ncol] = val;
        }
      }
    }
  }
}

// ---------------------------------------------------------------------------
// Flash attention: block = (64 queries) x (b,h); 128 threads = 4 waves,
// each wave owns 16 query rows. Key blocks of 64, online softmax.
// q,k: [b,h,s,dk] bf16.  vt: [b,h,dk,s] bf16 (pre-transposed by projection).
// ---------------------------------------------------------------------------
__global__ __launch_bounds__(128) void attn_kernel(const __bf16* __restrict__ q,
                                                   const __bf16* __restrict__ k,
                                                   const __bf16* __restrict__ vt,
                                                   const float* __restrict__ rel,
                                                   __bf16* __restrict__ o) {
  __shared__ __align__(16) __bf16 lK[64 * 64];        // [key][dk]
  __shared__ __align__(16) __bf16 lV[64 * 64];        // [dk][key]
  __shared__ __align__(16) __bf16 lP[4 * 16 * 64];    // per-wave P tile [row][key]
  __shared__ float relS[WIDTHc + 1];

  const int tid   = threadIdx.x;
  const int lane  = tid & 31;
  const int wv    = tid >> 5;
  const int lh    = lane & 15;
  const int khalf = lane >> 4;
  const int qBase = blockIdx.x * 64;
  const int h     = blockIdx.y;
  const int b     = blockIdx.z;

  if (tid <= WIDTHc) relS[tid] = rel[h * (WIDTHc + 1) + tid];

  const size_t headOff = (size_t)(b * Hc + h) * Sc * DKc;
  const __bf16* qh  = q  + headOff;
  const __bf16* kh  = k  + headOff;
  const __bf16* vth = vt + headOff;

  // Q fragments resident in registers (16 rows x 64 dk per wave)
  const int qrow = qBase + wv * 16 + lh;
  v16bf qf[2];
#pragma unroll
  for (int j = 0; j < 2; ++j)
    qf[j] = ld_frag(&qh[(size_t)qrow * 64 + j * 32 + khalf * 8],
                    &qh[(size_t)qrow * 64 + j * 32 + 16 + khalf * 8]);

  float mstate[8], lstate[8];
  v8f oacc[4];
#pragma unroll
  for (int r = 0; r < 8; ++r) { mstate[r] = -1e30f; lstate[r] = 0.f; }
#pragma unroll
  for (int c = 0; c < 4; ++c) oacc[c] = v8f_zero();

  __syncthreads();  // relS visible

  __bf16* pw = &lP[wv * 16 * 64];

  for (int kb = 0; kb < Sc; kb += 64) {
    // ---- stage K tile 64x64 [key][dk] + V tile 64x64 [dk][key] ----
#pragma unroll
    for (int i = 0; i < 4; ++i) {
      const int chunk = tid + i * 128;               // 512 chunks of 8 bf16
      const int key = chunk >> 3, dk = (chunk & 7) * 8;
      copy16_to_lds(&kh[(size_t)(kb + key) * 64 + dk], &lK[key * 64 + dk]);
    }
#pragma unroll
    for (int i = 0; i < 4; ++i) {
      const int chunk = tid + i * 128;
      const int dk = chunk >> 3, k8 = (chunk & 7) * 8;
      copy16_to_lds(&vth[(size_t)dk * Sc + kb + k8], &lV[dk * 64 + k8]);
    }
    // prefetch next tiles into cache (global_prefetch_b8)
    if (kb + 64 < Sc) {
      __builtin_prefetch(&kh[(size_t)(kb + 64 + (tid >> 1)) * 64 + (tid & 1) * 32], 0, 1);
      __builtin_prefetch(&vth[(size_t)(tid >> 1) * Sc + kb + 64 + (tid & 1) * 32], 0, 1);
    }
    async_copy_wait();
    __syncthreads();

    // ---- scores: 4 subtiles of 16 keys, K-depth 64 = 2 chained WMMAs ----
    v8f sacc[4];
#pragma unroll
    for (int t = 0; t < 4; ++t) {
      sacc[t] = v8f_zero();
      const int kcol = t * 16 + lh;
#pragma unroll
      for (int j = 0; j < 2; ++j) {
        v16bf kf = ld_frag(&lK[kcol * 64 + j * 32 + khalf * 8],
                           &lK[kcol * 64 + j * 32 + 16 + khalf * 8]);
        sacc[t] = __builtin_amdgcn_wmma_f32_16x16x32_bf16(
            false, qf[j], false, kf, (short)0, sacc[t], false, false);
      }
    }

    // ---- online softmax over 64 new keys (rows live in 16-lane halves) ----
#pragma unroll
    for (int r = 0; r < 8; ++r) {
      const int irow = qBase + wv * 16 + r + khalf * 8;
      const int prow = r + khalf * 8;
      float s[4];
      float rmax = -1e30f;
#pragma unroll
      for (int t = 0; t < 4; ++t) {
        const int jcol = kb + t * 16 + lh;
        const int d = irow - jcol;
        const float bv = (d >= 0 && d <= WIDTHc) ? relS[WIDTHc - d] : 0.f;
        s[t] = sacc[t][r] * SCALE_F + bv;
        rmax = fmaxf(rmax, s[t]);
      }
#pragma unroll
      for (int off = 1; off < 16; off <<= 1)
        rmax = fmaxf(rmax, __shfl_xor(rmax, off, 32));
      const float mnew = fmaxf(mstate[r], rmax);
      const float corr = __expf(mstate[r] - mnew);
      float rsum = 0.f;
#pragma unroll
      for (int t = 0; t < 4; ++t) {
        const float e = __expf(s[t] - mnew);
        pw[prow * 64 + t * 16 + lh] = (__bf16)e;   // C-layout -> A-layout via LDS
        rsum += e;
      }
#pragma unroll
      for (int off = 1; off < 16; off <<= 1)
        rsum += __shfl_xor(rsum, off, 32);
      lstate[r] = lstate[r] * corr + rsum;
      mstate[r] = mnew;
#pragma unroll
      for (int c = 0; c < 4; ++c) oacc[c][r] *= corr;
    }
    asm volatile("s_wait_dscnt 0" ::: "memory");  // own-wave LDS RAW fence

    // ---- O += P V  (K-depth 64 = 2 chained WMMAs per dk tile) ----
    v16bf pf[2];
#pragma unroll
    for (int j = 0; j < 2; ++j)
      pf[j] = ld_frag(&pw[lh * 64 + j * 32 + khalf * 8],
                      &pw[lh * 64 + j * 32 + 16 + khalf * 8]);
#pragma unroll
    for (int c = 0; c < 4; ++c) {
      const int dkc = c * 16 + lh;
#pragma unroll
      for (int j = 0; j < 2; ++j) {
        v16bf vf = ld_frag(&lV[dkc * 64 + j * 32 + khalf * 8],
                           &lV[dkc * 64 + j * 32 + 16 + khalf * 8]);
        oacc[c] = __builtin_amdgcn_wmma_f32_16x16x32_bf16(
            false, pf[j], false, vf, (short)0, oacc[c], false, false);
      }
    }
    __syncthreads();
  }

  // ---- normalize and write O as bf16 in [b, s, h*64+dk] ----
#pragma unroll
  for (int c = 0; c < 4; ++c) {
    const int col = h * 64 + c * 16 + lh;
#pragma unroll
    for (int r = 0; r < 8; ++r) {
      const int srow = qBase + wv * 16 + r + khalf * 8;
      const float val = oacc[c][r] / lstate[r];
      o[((size_t)b * Sc + srow) * Dc + col] = (__bf16)val;
    }
  }
}

// ---------------------------------------------------------------------------
extern "C" void kernel_launch(void* const* d_in, const int* in_sizes, int n_in,
                              void* d_out, int out_size, void* d_ws, size_t ws_size,
                              hipStream_t stream) {
  const float* query = (const float*)d_in[0];
  const float* key   = (const float*)d_in[1];
  const float* value = (const float*)d_in[2];
  const float* Wq = (const float*)d_in[3];
  const float* bq = (const float*)d_in[4];
  const float* Wk = (const float*)d_in[5];
  const float* bk = (const float*)d_in[6];
  const float* Wv = (const float*)d_in[7];
  const float* bv = (const float*)d_in[8];
  const float* Wo = (const float*)d_in[9];
  const float* bo = (const float*)d_in[10];
  const float* rel = (const float*)d_in[11];

  constexpr size_t HEAD_ELEMS = (size_t)Bc * Hc * Sc * DKc;  // 4,194,304
  __bf16* qws  = (__bf16*)d_ws;
  __bf16* kws  = qws + HEAD_ELEMS;
  __bf16* vtws = kws + HEAD_ELEMS;
  __bf16* ows  = vtws + HEAD_ELEMS;
  __bf16* wt   = ows + HEAD_ELEMS;    // 1M-elem bf16 scratch, reused per weight

  const dim3 gGemm(Dc / 128, (Bc * Sc) / 128);   // (8, 32)
  const dim3 gPrep(Dc / 32, Dc / 32);            // (32, 32)
  const dim3 blkGemm(256);
  const dim3 blkAttn(128);

  // serial stream ordering lets one wt buffer be reused per weight
  wprep_kernel<<<gPrep, blkGemm, 0, stream>>>(Wq, wt);
  gemm_kernel<false, 0><<<gGemm, blkGemm, 0, stream>>>(query, wt, bq, qws);
  wprep_kernel<<<gPrep, blkGemm, 0, stream>>>(Wk, wt);
  gemm_kernel<false, 0><<<gGemm, blkGemm, 0, stream>>>(key, wt, bk, kws);
  wprep_kernel<<<gPrep, blkGemm, 0, stream>>>(Wv, wt);
  gemm_kernel<false, 1><<<gGemm, blkGemm, 0, stream>>>(value, wt, bv, vtws);

  const dim3 gAttn(Sc / 64, Hc, Bc);             // (32, 16, 2)
  attn_kernel<<<gAttn, blkAttn, 0, stream>>>(qws, kws, vtws, rel, ows);

  wprep_kernel<<<gPrep, blkGemm, 0, stream>>>(Wo, wt);
  gemm_kernel<true, 2><<<gGemm, blkGemm, 0, stream>>>(ows, wt, bo, (float*)d_out);
}
